// ScaledDotProductAttention_33062658244999
// MI455X (gfx1250) — matrix operand order repaired and checked by reading, not compiled, not run on hardware
//
#include <hip/hip_runtime.h>
#include <hip/hip_bf16.h>
#include <stdint.h>

#define D_MODEL 512
#define SEQ     2048
#define NBATCH  8
#define NEGBIG  (-1.0e9f)

typedef __bf16 v16bf __attribute__((ext_vector_type(16)));
typedef float  v8f   __attribute__((ext_vector_type(8)));

union FragB { uint4 u[2]; v16bf v; };

static __device__ __forceinline__ unsigned short f2bf(float f) {
  union { float f; uint32_t u; } c; c.f = f;
  uint32_t r = (c.u + 0x7FFFu + ((c.u >> 16) & 1u)) >> 16;
  return (unsigned short)r;
}

// async 16-byte global -> LDS copy (ASYNCcnt-tracked, gfx1250)
static __device__ __forceinline__ void async_cp16(const void* g, void* l) {
  uint32_t lds_off = (uint32_t)(uintptr_t)l;   // low 32 bits = LDS byte offset
  asm volatile("global_load_async_to_lds_b128 %0, %1, off"
               :: "v"(lds_off), "v"((uint64_t)(uintptr_t)g)
               : "memory");
}

// ---------------------------------------------------------------------------
// Kernel 0: fp32 -> bf16 conversion (grid-stride)
// ---------------------------------------------------------------------------
__global__ void cvt_f32_bf16(const float* __restrict__ src,
                             unsigned short* __restrict__ dst, int n) {
  int i = blockIdx.x * blockDim.x + threadIdx.x;
  int stride = gridDim.x * blockDim.x;
  for (; i < n; i += stride) dst[i] = f2bf(src[i]);
}

// ---------------------------------------------------------------------------
// Kernel 1: fused QKV projection.  q[b,s,e] = sum_d x[b,s,d]*W[e,d] + bias[e]
// One wave -> one 16(M) x 64(N) tile of one of the 3 output matrices.
// Q,K stored row-major bf16 [b*S + s][e]; V stored transposed vt[b][e][s].
// ---------------------------------------------------------------------------
__global__ __launch_bounds__(256) void qkv_gemm(
    const unsigned short* __restrict__ xb,
    const unsigned short* __restrict__ wb,
    const float* __restrict__ bq, const float* __restrict__ bk,
    const float* __restrict__ bv,
    unsigned short* __restrict__ qb, unsigned short* __restrict__ kb,
    unsigned short* __restrict__ vt) {
  const int lane  = threadIdx.x & 31;
  const int wave  = threadIdx.x >> 5;
  const int tile  = blockIdx.x * 8 + wave;
  const int mat   = tile >> 13;
  const int rem   = tile & 8191;
  const int m0    = (rem >> 3) << 4;
  const int n0    = (rem & 7) << 6;
  const int lhalf = lane >> 4;
  const int lmod  = lane & 15;

  v8f acc[4] = {};
  const unsigned short* w    = wb + (size_t)mat * (D_MODEL * D_MODEL);
  const unsigned short* arow = xb + (size_t)(m0 + lmod) * D_MODEL + (lhalf ? 8 : 0);

  for (int k0 = 0; k0 < D_MODEL; k0 += 32) {
    FragB a;
    a.u[0] = *(const uint4*)(arow + k0);
    a.u[1] = *(const uint4*)(arow + k0 + 16);
#pragma unroll
    for (int nt = 0; nt < 4; ++nt) {
      const unsigned short* bp =
          w + (size_t)(n0 + nt * 16 + lmod) * D_MODEL + k0 + (lhalf ? 8 : 0);
      FragB bm;
      bm.u[0] = *(const uint4*)(bp);
      bm.u[1] = *(const uint4*)(bp + 16);
      acc[nt] = __builtin_amdgcn_wmma_f32_16x16x32_bf16(
          false, a.v, false, bm.v, (short)0, acc[nt], false, false);
    }
  }

  const float* bias = (mat == 0) ? bq : (mat == 1) ? bk : bv;
  const int b  = m0 >> 11;
  const int s0 = m0 & 2047;

#pragma unroll
  for (int nt = 0; nt < 4; ++nt) {
    const int e = n0 + nt * 16 + lmod;
    const float bs = bias[e];
    if (mat == 2) {
      unsigned short pk[8];
#pragma unroll
      for (int r = 0; r < 8; ++r) pk[r] = f2bf(acc[nt][r] + bs);
      *(uint4*)(vt + ((size_t)(b * D_MODEL + e)) * SEQ + s0 + 8 * lhalf) =
          *(const uint4*)pk;
    } else {
      unsigned short* dst = (mat == 0) ? qb : kb;
#pragma unroll
      for (int r = 0; r < 8; ++r) {
        const int row = m0 + r + 8 * lhalf;
        dst[(size_t)row * D_MODEL + e] = f2bf(acc[nt][r] + bs);
      }
    }
  }
}

// ---------------------------------------------------------------------------
// Kernel 2: masked attention, flash-style online softmax.
// 8 waves/block share DOUBLE-BUFFERED async-staged K/V blocks in LDS;
// Q A-frags persistent in VGPRs.  ~292 KB LDS (<320 KB WG limit).
// ---------------------------------------------------------------------------
#define AW 8
#define KROW (D_MODEL + 16)   // +32B row pad
#define VROW (64 + 8)         // +16B row pad
__global__ __launch_bounds__(32 * AW) void attn_fwd(
    const unsigned short* __restrict__ qb,
    const unsigned short* __restrict__ kbuf,
    const unsigned short* __restrict__ vt,
    const int* __restrict__ lens,
    float* __restrict__ out) {
  __shared__ __align__(16) unsigned short Ks[2][64][KROW];      // 132 KB
  __shared__ __align__(16) unsigned short Vs[2][D_MODEL][VROW]; // 144 KB
  __shared__ __align__(16) unsigned short Ps[AW][16 * 64];      //  16 KB

  const int tid   = threadIdx.x;
  const int lane  = tid & 31;
  const int wave  = tid >> 5;
  const int tile  = blockIdx.x * AW + wave;     // 0 .. B*S/16-1
  const int b     = tile >> 7;                  // same batch for whole block
  const int s0    = (tile & 127) << 4;
  const int lhalf = lane >> 4;
  const int lmod  = lane & 15;
  const int L     = lens[b];
  const float inv_scale = 0.044194173824159216f;  // 1/sqrt(512)

  const unsigned short* kbase = kbuf + (size_t)b * SEQ * D_MODEL;
  const unsigned short* vbase = vt + (size_t)b * D_MODEL * SEQ;

  // issue async copies for one 64-key K/V block into buffer `buf`
  auto stage = [&](int kb0, int buf) {
    for (int idx = tid; idx < 64 * 64; idx += 32 * AW) {       // K: 4096 uint4
      const int key = idx >> 6, c = (idx & 63) << 3;
      async_cp16(kbase + (size_t)(kb0 + key) * D_MODEL + c, &Ks[buf][key][c]);
    }
    for (int idx = tid; idx < D_MODEL * 8; idx += 32 * AW) {   // V: 4096 uint4
      const int e = idx >> 3, c = (idx & 7) << 3;
      async_cp16(vbase + (size_t)e * SEQ + kb0 + c, &Vs[buf][e][c]);
    }
  };

  // ---- Q tile: 16 pre-built A-fragments held in VGPRs (128 VGPRs) ----------
  FragB qf[16];
  {
    const unsigned short* qrow =
        qb + ((size_t)b * SEQ + s0 + lmod) * D_MODEL + (lhalf ? 8 : 0);
#pragma unroll
    for (int k = 0; k < 16; ++k) {
      qf[k].u[0] = *(const uint4*)(qrow + k * 32);
      qf[k].u[1] = *(const uint4*)(qrow + k * 32 + 16);
    }
  }

  float m_run[8], l_run[8];
#pragma unroll
  for (int r = 0; r < 8; ++r) { m_run[r] = -__builtin_inff(); l_run[r] = 0.f; }
  v8f accO[32] = {};

  stage(0, 0);                                   // prologue

  for (int it = 0; it < SEQ / 64; ++it) {
    const int kb0 = it * 64;
    const int cur = it & 1;

    asm volatile("s_wait_asynccnt 0x0" ::: "memory");  // my copies for `cur` done
    __syncthreads();                                   // everyone's copies done
    if (it + 1 < SEQ / 64) stage(kb0 + 64, cur ^ 1);   // prefetch next block

    // ---- scores: S = Q K^T for 16 x 64 block --------------------------------
    v8f sc[4] = {};
#pragma unroll
    for (int k = 0; k < 16; ++k) {
      const int koff = k * 32 + (lhalf ? 8 : 0);
#pragma unroll
      for (int nt = 0; nt < 4; ++nt) {
        const unsigned short* kp = &Ks[cur][nt * 16 + lmod][koff];
        FragB bm;
        bm.u[0] = *(const uint4*)kp;
        bm.u[1] = *(const uint4*)(kp + 16);
        sc[nt] = __builtin_amdgcn_wmma_f32_16x16x32_bf16(
            false, qf[k].v, false, bm.v, (short)0, sc[nt], false, false);
      }
    }
    // ---- scale + mask -------------------------------------------------------
#pragma unroll
    for (int nt = 0; nt < 4; ++nt) {
      const int key = kb0 + nt * 16 + lmod;
      const bool kok = key < L;
#pragma unroll
      for (int r = 0; r < 8; ++r) {
        const int qidx = s0 + r + 8 * lhalf;
        const float sv = sc[nt][r] * inv_scale;
        sc[nt][r] = (kok && qidx < L) ? sv : NEGBIG;
      }
    }
    // ---- online softmax (row r lives on 16 lanes of a half-wave) -----------
#pragma unroll
    for (int r = 0; r < 8; ++r) {
      float mx = fmaxf(fmaxf(sc[0][r], sc[1][r]), fmaxf(sc[2][r], sc[3][r]));
      mx = fmaxf(mx, __shfl_xor(mx, 1, 32));
      mx = fmaxf(mx, __shfl_xor(mx, 2, 32));
      mx = fmaxf(mx, __shfl_xor(mx, 4, 32));
      mx = fmaxf(mx, __shfl_xor(mx, 8, 32));
      const float mnew = fmaxf(m_run[r], mx);
      const float corr = __expf(m_run[r] - mnew);
      m_run[r] = mnew;
      float rs = 0.f;
#pragma unroll
      for (int nt = 0; nt < 4; ++nt) {
        const float e = __expf(sc[nt][r] - mnew);
        sc[nt][r] = e;
        rs += e;
      }
      rs += __shfl_xor(rs, 1, 32);
      rs += __shfl_xor(rs, 2, 32);
      rs += __shfl_xor(rs, 4, 32);
      rs += __shfl_xor(rs, 8, 32);
      l_run[r] = l_run[r] * corr + rs;
#pragma unroll
      for (int dt = 0; dt < 32; ++dt) accO[dt][r] *= corr;
    }
    // ---- P: C-frag layout -> A-frag layout via (wave-private) LDS ----------
#pragma unroll
    for (int nt = 0; nt < 4; ++nt)
#pragma unroll
      for (int r = 0; r < 8; ++r) {
        const int row = r + 8 * lhalf;
        Ps[wave][row * 64 + nt * 16 + lmod] = f2bf(sc[nt][r]);
      }
    asm volatile("s_wait_dscnt 0x0" ::: "memory");
    // ---- O += P @ V  (contract 64 keys = 2 k-steps) -------------------------
#pragma unroll
    for (int ks = 0; ks < 2; ++ks) {
      FragB a;
      const unsigned short* prow = Ps[wave] + lmod * 64 + ks * 32 + (lhalf ? 8 : 0);
      a.u[0] = *(const uint4*)(prow);
      a.u[1] = *(const uint4*)(prow + 16);
#pragma unroll
      for (int dt = 0; dt < 32; ++dt) {
        const unsigned short* vp = &Vs[cur][dt * 16 + lmod][ks * 32 + (lhalf ? 8 : 0)];
        FragB bm;
        bm.u[0] = *(const uint4*)vp;
        bm.u[1] = *(const uint4*)(vp + 16);
        accO[dt] = __builtin_amdgcn_wmma_f32_16x16x32_bf16(
            false, a.v, false, bm.v, (short)0, accO[dt], false, false);
      }
    }
  }

  // ---- normalize and store fp32 output -------------------------------------
  float inv_l[8];
#pragma unroll
  for (int r = 0; r < 8; ++r) inv_l[r] = 1.0f / l_run[r];
  float* obase = out + ((size_t)b * SEQ + s0) * D_MODEL;
#pragma unroll
  for (int dt = 0; dt < 32; ++dt) {
    const int e = dt * 16 + lmod;
#pragma unroll
    for (int r = 0; r < 8; ++r) {
      const int row = r + 8 * lhalf;
      obase[(size_t)row * D_MODEL + e] = accO[dt][r] * inv_l[r];
    }
  }
}

// ---------------------------------------------------------------------------
extern "C" void kernel_launch(void* const* d_in, const int* in_sizes, int n_in,
                              void* d_out, int out_size, void* d_ws,
                              size_t ws_size, hipStream_t stream) {
  const float* x  = (const float*)d_in[0];
  const int* lens = (const int*)d_in[1];
  const float* Wq = (const float*)d_in[2];
  const float* bq = (const float*)d_in[3];
  const float* Wk = (const float*)d_in[4];
  const float* bk = (const float*)d_in[5];
  const float* Wv = (const float*)d_in[6];
  const float* bv = (const float*)d_in[7];
  float* out = (float*)d_out;

  char* ws = (char*)d_ws;
  unsigned short* xb  = (unsigned short*)(ws);
  unsigned short* wb  = (unsigned short*)(ws + (size_t)16 * 1048576);
  unsigned short* qb  = (unsigned short*)(ws + (size_t)18 * 1048576);
  unsigned short* kb  = (unsigned short*)(ws + (size_t)34 * 1048576);
  unsigned short* vtb = (unsigned short*)(ws + (size_t)50 * 1048576);

  const int NX = NBATCH * SEQ * D_MODEL;
  const int NW = D_MODEL * D_MODEL;

  cvt_f32_bf16<<<1024, 256, 0, stream>>>(x,  xb, NX);
  cvt_f32_bf16<<<256, 256, 0, stream>>>(Wq, wb + 0 * (size_t)NW, NW);
  cvt_f32_bf16<<<256, 256, 0, stream>>>(Wk, wb + 1 * (size_t)NW, NW);
  cvt_f32_bf16<<<256, 256, 0, stream>>>(Wv, wb + 2 * (size_t)NW, NW);

  qkv_gemm<<<3072, 256, 0, stream>>>(xb, wb, bq, bk, bv, qb, kb, vtb);

  // B*S/16 = 1024 q-tiles, AW waves per block
  attn_fwd<<<1024 / AW, 32 * AW, 0, stream>>>(qb, kb, vtb, lens, out);
}
